// RelationBranch_58849641889963
// MI455X (gfx1250) — matrix operand order repaired and checked by reading, not compiled, not run on hardware
//
#include <hip/hip_runtime.h>
#include <hip/hip_bf16.h>
#include <math.h>

// ---------------------------------------------------------------------------
// Types / helpers
// ---------------------------------------------------------------------------
typedef unsigned short u16;
typedef __attribute__((ext_vector_type(16))) __bf16 bf16x16;
typedef __attribute__((ext_vector_type(8)))  __bf16 bf16x8;
typedef __attribute__((ext_vector_type(8)))  float  f32x8;
typedef __attribute__((ext_vector_type(8)))  unsigned short u16x8;

__device__ __forceinline__ u16 f2b(float f) {  // fp32 -> bf16, round-to-nearest-even
  unsigned int u = __float_as_uint(f);
  u += 0x7fffu + ((u >> 16) & 1u);
  return (u16)(u >> 16);
}

// Build a 16-element bf16 fragment from two contiguous 8-element (16B) chunks.
__device__ __forceinline__ bf16x16 mkfrag(const u16* p0, const u16* p1) {
  u16x8 a = *(const u16x8*)p0;
  u16x8 b = *(const u16x8*)p1;
  bf16x8 x = __builtin_bit_cast(bf16x8, a);
  bf16x8 y = __builtin_bit_cast(bf16x8, b);
  return __builtin_shufflevector(x, y, 0,1,2,3,4,5,6,7,8,9,10,11,12,13,14,15);
}

__device__ __forceinline__ f32x8 wmma_bf16(bf16x16 a, bf16x16 b, f32x8 c) {
  return __builtin_amdgcn_wmma_f32_16x16x32_bf16(false, a, false, b, (short)0, c,
                                                 false, false);
}

// Wave-private LDS ordering: stores above complete before loads below.
__device__ __forceinline__ void wait_ds0() {
  asm volatile("s_wait_dscnt 0x0" ::: "memory");
}

// ---- CDNA5 async global->LDS (guarded: compiles either way) ----------------
// Builtin signature (from toolchain diagnostic): param0 = addrspace(1) v4i*,
// param1 = addrspace(3) v4i*, then two immediate ints (offset, cpol).
#if __has_builtin(__builtin_amdgcn_global_load_async_to_lds_b128)
#define HAVE_ASYNC_LDS 1
typedef int v4i __attribute__((vector_size(16)));
typedef __attribute__((address_space(1))) v4i g_v4i;
typedef __attribute__((address_space(3))) v4i l_v4i;
__device__ __forceinline__ void async_b128(const void* src, void* dst_lds) {
  __builtin_amdgcn_global_load_async_to_lds_b128(
      (g_v4i*)(unsigned long long)src,                        // flat==global addr
      (l_v4i*)(unsigned int)(unsigned long long)dst_lds,      // flat->LDS low 32b
      0, 0);
}
__device__ __forceinline__ void wait_async0() {
  asm volatile("s_wait_asynccnt 0x0" ::: "memory");
}
#else
__device__ __forceinline__ void wait_async0() {}
#endif

// ---------------------------------------------------------------------------
// Generic bf16 GEMM: C[M,N] = A[M,K] @ B[K,N] (+bias, +relu).
// Block: 256 thr = 8 waves, tile 128x128x32, double-buffered LDS staging
// (one barrier per K-step).  Wave tile 32x64 (2x4 WMMAs).
// flags: bit0 = add bias, bit1 = relu, bit2 = store Cb transposed [N][M]
// Cf (fp32) and Cb (bf16) are each optional (nullptr to skip).
// ---------------------------------------------------------------------------
#define TM 128
#define TN 128
#define TK 32

__global__ __launch_bounds__(256)
void gemm_bf16(const u16* __restrict__ A, const u16* __restrict__ B,
               const float* __restrict__ bias, float* __restrict__ Cf,
               u16* __restrict__ Cb, int M, int N, int K, int flags)
{
  __shared__ __align__(16) u16 As[2 * TM * TK];   // [buf][m][k] (k contiguous)
  __shared__ __align__(16) u16 Bs[2 * TN * TK];   // [buf][n][k] (transposed)

  const int tid  = threadIdx.x;
  const int lane = tid & 31, wave = tid >> 5;
  const int half = lane >> 4, l16 = lane & 15;
  const int waveM = wave >> 1, waveN = wave & 1;
  const int bm = blockIdx.y * TM, bn = blockIdx.x * TN;

  f32x8 acc[2][4];
  const f32x8 zero8 = {0.f,0.f,0.f,0.f,0.f,0.f,0.f,0.f};
  for (int i = 0; i < 2; ++i)
    for (int j = 0; j < 4; ++j) acc[i][j] = zero8;

  const int arow = tid >> 1, aseg = tid & 1;              // A staging
  const int bn2  = (tid & 63) * 2, bk0 = (tid >> 6) * 8;  // B staging

  // Stage one 128x32 A tile + 32x128 B tile (transposed) into buffer `buf`.
  auto stage = [&](int k0, int buf) {
    u16* Asb = As + buf * (TM * TK);
    u16* Bsb = Bs + buf * (TN * TK);
    { // A tile
      u16* dst = Asb + arow * TK + aseg * 16;
#ifdef HAVE_ASYNC_LDS
      if (bm + arow < M) {
        const u16* src = A + (size_t)(bm + arow) * K + k0 + aseg * 16;
        async_b128(src, dst);
        async_b128(src + 8, dst + 8);
      } else {
        u16x8 z = {0,0,0,0,0,0,0,0};
        *(u16x8*)dst = z; *(u16x8*)(dst + 8) = z;
      }
#else
      u16x8 v0 = {0,0,0,0,0,0,0,0}, v1 = v0;
      if (bm + arow < M) {
        const u16* src = A + (size_t)(bm + arow) * K + k0 + aseg * 16;
        v0 = *(const u16x8*)src;
        v1 = *(const u16x8*)(src + 8);
      }
      *(u16x8*)dst = v0; *(u16x8*)(dst + 8) = v1;
#endif
      if (bm + arow < M && k0 + 2 * TK < K)   // prefetch tile after next
        __builtin_prefetch(A + (size_t)(bm + arow) * K + k0 + 2 * TK + aseg * 16, 0, 1);
    }
    { // B tile, transposed: Bsb[n][k] = B[k0+k][bn+n]
      for (int kk = 0; kk < 8; ++kk) {
        unsigned int w =
            *(const unsigned int*)(B + (size_t)(k0 + bk0 + kk) * N + bn + bn2);
        Bsb[bn2 * TK + bk0 + kk]       = (u16)(w & 0xffffu);
        Bsb[(bn2 + 1) * TK + bk0 + kk] = (u16)(w >> 16);
      }
      if (k0 + 2 * TK < K)
        __builtin_prefetch(B + (size_t)(k0 + 2 * TK + bk0) * N + bn + bn2, 0, 1);
    }
  };

  int cur = 0;
  stage(0, 0);
  wait_async0();
  __syncthreads();

  for (int k0 = 0; k0 < K; k0 += TK) {
    if (k0 + TK < K) stage(k0 + TK, cur ^ 1);   // overlap next tile with compute

    const u16* Asb = As + cur * (TM * TK);
    const u16* Bsb = Bs + cur * (TN * TK);
    bf16x16 af[2], bfr[4];
    for (int mt = 0; mt < 2; ++mt) {
      // A frag: lanes<16 hold K0..15, lanes>=16 K16..31 (contiguous)
      const u16* p = Asb + (waveM * 32 + mt * 16 + l16) * TK;
      af[mt] = mkfrag(p + half * 16, p + half * 16 + 8);
    }
    for (int nt = 0; nt < 4; ++nt) {
      // B frag: lanes<16 hold K0..15, lanes>=16 K16..31 (contiguous)
      const u16* p = Bsb + (waveN * 64 + nt * 16 + l16) * TK + half * 16;
      bfr[nt] = mkfrag(p, p + 8);
    }
    for (int mt = 0; mt < 2; ++mt)
      for (int nt = 0; nt < 4; ++nt)
        acc[mt][nt] = wmma_bf16(af[mt], bfr[nt], acc[mt][nt]);

    wait_async0();       // next-tile async writes visible before barrier
    __syncthreads();     // single barrier per K-step
    cur ^= 1;
  }

  // Epilogue: C/D layout -> row = j + 8*half, col = l16 within each 16x16 tile
  for (int mt = 0; mt < 2; ++mt)
    for (int nt = 0; nt < 4; ++nt) {
      int cn = bn + waveN * 64 + nt * 16 + l16;
      float bv = (flags & 1) ? bias[cn] : 0.f;
      for (int j = 0; j < 8; ++j) {
        int r = bm + waveM * 32 + mt * 16 + j + 8 * half;
        if (r < M) {
          float v = acc[mt][nt][j] + bv;
          if (flags & 2) v = v > 0.f ? v : 0.f;
          if (Cf) Cf[(size_t)r * N + cn] = v;
          if (Cb) {
            if (flags & 4) Cb[(size_t)cn * M + r] = f2b(v);  // transposed [N][M]
            else           Cb[(size_t)r * N + cn] = f2b(v);
          }
        }
      }
    }
}

// ---------------------------------------------------------------------------
// Flash cross-attention. 1 block = 16 queries x all 8 heads (1 wave/head).
// Q,K bf16 [rows][256]; V pre-transposed Vt bf16 [256][Mkeys].
// Per 32-key block: S via 2 WMMAs, online softmax, P transposed through
// wave-private LDS (ordered by s_wait_dscnt, no workgroup barrier),
// PV via 2 WMMAs with contiguous Vt fragments.  Output bf16 [P][256].
// ---------------------------------------------------------------------------
__global__ __launch_bounds__(256)
void attention_fa(const u16* __restrict__ Qb, const u16* __restrict__ Kb,
                  const u16* __restrict__ Vt, const unsigned char* __restrict__ mmask,
                  u16* __restrict__ Ob, float scale, int Mkeys)
{
  __shared__ __align__(16) u16 Plds[8 * 16 * 32];   // per-wave 16x32 bf16 tile

  const int tid  = threadIdx.x;
  const int wave = tid >> 5, lane = tid & 31;
  const int half = lane >> 4, l16 = lane & 15;
  const int h  = wave;                // head
  const int p0 = blockIdx.x * 16;     // query tile base (P = 2016 = 126*16 exact)

  const u16* qptr = Qb + (size_t)(p0 + l16) * 256 + h * 32;
  bf16x16 qf = mkfrag(qptr + half * 16, qptr + half * 16 + 8);

  const f32x8 zero8 = {0.f,0.f,0.f,0.f,0.f,0.f,0.f,0.f};
  f32x8 o0 = zero8, o1 = zero8;
  float rmax[8], rsum[8];
  for (int j = 0; j < 8; ++j) { rmax[j] = -3.0e38f; rsum[j] = 0.f; }

  u16* pl = Plds + wave * 512;
  const u16* vrow0 = Vt + (size_t)(h * 32 + l16) * (size_t)Mkeys;
  const u16* vrow1 = Vt + (size_t)(h * 32 + 16 + l16) * (size_t)Mkeys;

  for (int kb = 0; kb < Mkeys; kb += 32) {
    // K fragments: B[k][n] = K[key n][dim k] -> contiguous 32B per lane
    const u16* kp0 = Kb + (size_t)(kb + l16) * 256 + h * 32 + half * 16;
    const u16* kp1 = Kb + (size_t)(kb + 16 + l16) * 256 + h * 32 + half * 16;
    bf16x16 kf0 = mkfrag(kp0, kp0 + 8);
    bf16x16 kf1 = mkfrag(kp1, kp1 + 8);

    // prefetch next key block (K rows + Vt runs) into the cache hierarchy
    if (kb + 32 < Mkeys) {
      __builtin_prefetch(Kb + (size_t)(kb + 32 + l16) * 256 + h * 32, 0, 1);
      __builtin_prefetch(Kb + (size_t)(kb + 48 + l16) * 256 + h * 32, 0, 1);
      __builtin_prefetch(vrow0 + kb + 32, 0, 1);
      __builtin_prefetch(vrow1 + kb + 32, 0, 1);
    }

    f32x8 s0 = wmma_bf16(qf, kf0, zero8);
    f32x8 s1 = wmma_bf16(qf, kf1, zero8);

    const bool msk0 = mmask[kb + l16] != 0;          // this lane's key column
    const bool msk1 = mmask[kb + 16 + l16] != 0;

    float p0v[8], p1v[8];
    for (int j = 0; j < 8; ++j) {
      float a = s0[j] * scale; if (msk0) a = -1e9f;
      float b = s1[j] * scale; if (msk1) b = -1e9f;
      p0v[j] = a; p1v[j] = b;
    }
    // online softmax: row reductions across the 16 lanes of this half
    for (int j = 0; j < 8; ++j) {
      float t = fmaxf(p0v[j], p1v[j]);
      for (int m = 1; m < 16; m <<= 1) t = fmaxf(t, __shfl_xor(t, m, 32));
      float newm = fmaxf(rmax[j], t);
      float corr = __expf(rmax[j] - newm);
      rmax[j] = newm;
      float e0 = __expf(p0v[j] - newm);
      float e1 = __expf(p1v[j] - newm);
      p0v[j] = e0; p1v[j] = e1;
      float ssum = e0 + e1;
      for (int m = 1; m < 16; m <<= 1) ssum += __shfl_xor(ssum, m, 32);
      rsum[j] = rsum[j] * corr + ssum;
      o0[j] *= corr; o1[j] *= corr;
    }
    // transpose P (C layout -> [query][key] in LDS, bf16); wave-private tile,
    // so only DS-counter ordering is needed (no workgroup barrier).
    for (int j = 0; j < 8; ++j) {
      int m = j + 8 * half;
      pl[m * 32 + l16]      = f2b(p0v[j]);
      pl[m * 32 + 16 + l16] = f2b(p1v[j]);
    }
    wait_ds0();
    const u16* pp = pl + l16 * 32;
    bf16x16 pf = mkfrag(pp + half * 16, pp + half * 16 + 8);

    // V fragments from transposed Vt: contiguous 32B per lane
    const u16* vp0 = vrow0 + kb + half * 16;
    const u16* vp1 = vrow1 + kb + half * 16;
    bf16x16 vf0 = mkfrag(vp0, vp0 + 8);
    bf16x16 vf1 = mkfrag(vp1, vp1 + 8);

    o0 = wmma_bf16(pf, vf0, o0);
    o1 = wmma_bf16(pf, vf1, o1);
  }

  for (int j = 0; j < 8; ++j) {
    float inv = 1.f / rsum[j];
    size_t r = (size_t)(p0 + j + 8 * half) * 256 + h * 32;
    Ob[r + l16]      = f2b(o0[j] * inv);
    Ob[r + 16 + l16] = f2b(o1[j] * inv);
  }
}

// ---------------------------------------------------------------------------
// Residual + LayerNorm (one block per row of 256), dual fp32/bf16 output.
// ---------------------------------------------------------------------------
__global__ __launch_bounds__(256)
void ln_residual(const float* __restrict__ x, const float* __restrict__ rr,
                 const float* __restrict__ g, const float* __restrict__ b,
                 float* __restrict__ outf, u16* __restrict__ outb)
{
  __shared__ float red[256];
  const int row = blockIdx.x, t = threadIdx.x;
  float v = x[(size_t)row * 256 + t] + rr[(size_t)row * 256 + t];
  red[t] = v; __syncthreads();
  for (int s = 128; s > 0; s >>= 1) { if (t < s) red[t] += red[t + s]; __syncthreads(); }
  float mean = red[0] * (1.f / 256.f);
  __syncthreads();
  float d = v - mean;
  red[t] = d * d; __syncthreads();
  for (int s = 128; s > 0; s >>= 1) { if (t < s) red[t] += red[t + s]; __syncthreads(); }
  float var = red[0] * (1.f / 256.f);
  float y = d * rsqrtf(var + 1e-5f) * g[t] + b[t];
  outf[(size_t)row * 256 + t] = y;
  outb[(size_t)row * 256 + t] = f2b(y);
}

// ---------------------------------------------------------------------------
// Pair construction: p -> (row,col) of triu(k=1), concat bf16 features + refpts.
// ---------------------------------------------------------------------------
__device__ __forceinline__ void pair_rc(int p, int& r, int& c) {
  int rr = 0, cum = 0;
  while (cum + (63 - rr) <= p) { cum += 63 - rr; ++rr; }
  r = rr; c = rr + 1 + (p - cum);
}

__global__ __launch_bounds__(256)
void build_pairs(const float* __restrict__ h, const float* __restrict__ refs,
                 u16* __restrict__ pair, float* __restrict__ refpts)
{
  const int p = blockIdx.x, t = threadIdx.x;
  int r, c; pair_rc(p, r, c);
  pair[(size_t)p * 512 + t]       = f2b(h[r * 256 + t]);
  pair[(size_t)p * 512 + 256 + t] = f2b(h[c * 256 + t]);
  if (t == 0) {
    refpts[p * 2 + 0] = 0.5f * (refs[r * 4 + 0] + refs[c * 4 + 0]);
    refpts[p * 2 + 1] = 0.5f * (refs[r * 4 + 1] + refs[c * 4 + 1]);
  }
}

__global__ __launch_bounds__(256)
void add_ref_cvt(float* __restrict__ q, const float* __restrict__ refpts,
                 const float* __restrict__ Wref, u16* __restrict__ qb)
{
  const int p = blockIdx.x, e = threadIdx.x;
  float v = q[(size_t)p * 256 + e] + refpts[p * 2] * Wref[e]
          + refpts[p * 2 + 1] * Wref[256 + e];
  q[(size_t)p * 256 + e]  = v;
  qb[(size_t)p * 256 + e] = f2b(v);
}

// ---------------------------------------------------------------------------
// Contrastive scores + symmetric scatter into [64,64,32].
// ---------------------------------------------------------------------------
__global__ void scores_scatter(const float* __restrict__ q, const float* __restrict__ text,
                               const unsigned char* __restrict__ tmask,
                               const float* __restrict__ log_scale,
                               float* __restrict__ out)
{
  const int p = blockIdx.x, t = threadIdx.x;   // 32 threads
  int r, c; pair_rc(p, r, c);
  float s = 0.f;
  const float* qp = q + (size_t)p * 256;
  const float* tp = text + (size_t)t * 256;
  for (int e = 0; e < 256; ++e) s += qp[e] * tp[e];
  s *= __expf(log_scale[0]);
  if (!tmask[t]) s = -__builtin_inff();
  out[((size_t)r * 64 + c) * 32 + t] = s;
  out[((size_t)c * 64 + r) * 32 + t] = s;
}

__global__ void zero_f32(float* __restrict__ p, int n) {
  int i = blockIdx.x * 256 + threadIdx.x;
  if (i < n) p[i] = 0.f;
}

__global__ void cvt_f32_bf16(const float* __restrict__ s, u16* __restrict__ d, int n) {
  int i = blockIdx.x * 256 + threadIdx.x;
  if (i < n) d[i] = f2b(s[i]);
}

// ---------------------------------------------------------------------------
// Host orchestration
// ---------------------------------------------------------------------------
static inline size_t alignup(size_t x) { return (x + 255) & ~(size_t)255; }

extern "C" void kernel_launch(void* const* d_in, const int* in_sizes, int n_in,
                              void* d_out, int out_size, void* d_ws, size_t ws_size,
                              hipStream_t stream) {
  const float* hidden = (const float*)d_in[0];   // [64,256]
  const float* memory = (const float*)d_in[1];   // [4096,256]
  const float* refs   = (const float*)d_in[2];   // [64,4]
  const float* mtext  = (const float*)d_in[3];   // [32,256]
  const unsigned char* mmask = (const unsigned char*)d_in[4];  // [4096]
  const unsigned char* tmask = (const unsigned char*)d_in[5];  // [32]
  const float* W_pair = (const float*)d_in[6];
  const float* b_pair = (const float*)d_in[7];
  const float* W_mem  = (const float*)d_in[8];
  const float* b_mem  = (const float*)d_in[9];
  const float* W_text = (const float*)d_in[10];
  const float* b_text = (const float*)d_in[11];
  const float* W_ref  = (const float*)d_in[12];
  const float* Wq = (const float*)d_in[13]; const float* bq = (const float*)d_in[14];
  const float* Wk = (const float*)d_in[15]; const float* bk = (const float*)d_in[16];
  const float* Wv = (const float*)d_in[17]; const float* bv = (const float*)d_in[18];
  const float* Wo = (const float*)d_in[19]; const float* bo = (const float*)d_in[20];
  const float* ln1_g = (const float*)d_in[21]; const float* ln1_b = (const float*)d_in[22];
  const float* ln2_g = (const float*)d_in[23]; const float* ln2_b = (const float*)d_in[24];
  const float* W_ff1 = (const float*)d_in[25]; const float* b_ff1 = (const float*)d_in[26];
  const float* W_ff2 = (const float*)d_in[27]; const float* b_ff2 = (const float*)d_in[28];
  const float* log_scale = (const float*)d_in[29];

  const int P = 2016, E = 256, M = 4096, FF = 1024;

  // workspace carve
  char* w = (char*)d_ws; size_t off = 0;
  auto take = [&](size_t bytes) { char* p = w + off; off = alignup(off + bytes); return p; };
  u16* Wmem_b  = (u16*)take((size_t)E * E * 2);
  u16* Wtext_b = (u16*)take((size_t)E * E * 2);
  u16* Wpair_b = (u16*)take((size_t)2 * E * E * 2);
  u16* Wq_b = (u16*)take((size_t)4 * E * E * 2);
  u16* Wk_b = (u16*)take((size_t)4 * E * E * 2);
  u16* Wv_b = (u16*)take((size_t)4 * E * E * 2);
  u16* Wo_b = (u16*)take((size_t)4 * E * E * 2);
  u16* Wf1_b = (u16*)take((size_t)4 * E * FF * 2);
  u16* Wf2_b = (u16*)take((size_t)4 * FF * E * 2);
  u16* mem_b    = (u16*)take((size_t)M * E * 2);
  u16* memp_b   = (u16*)take((size_t)M * E * 2);
  u16* textin_b = (u16*)take((size_t)32 * E * 2);
  u16* K_b  = (u16*)take((size_t)M * E * 2);
  u16* Vt_b = (u16*)take((size_t)E * M * 2);      // V transposed [E][M]
  u16* pair_b = (u16*)take((size_t)P * 2 * E * 2);
  float* refpts = (float*)take((size_t)P * 2 * 4);
  float* q_f  = (float*)take((size_t)P * E * 4);
  u16* q_b    = (u16*)take((size_t)P * E * 2);
  u16* qh_b   = (u16*)take((size_t)P * E * 2);
  u16* attn_b = (u16*)take((size_t)P * E * 2);
  u16* f1_b   = (u16*)take((size_t)P * FF * 2);
  float* tmp2_f = (float*)take((size_t)P * E * 4);
  float* text_f = (float*)take((size_t)32 * E * 4);
  (void)ws_size; (void)in_sizes; (void)n_in; (void)out_size;

  auto cvt = [&](const float* s, u16* d, int n) {
    cvt_f32_bf16<<<(n + 255) / 256, 256, 0, stream>>>(s, d, n);
  };
  auto gemm = [&](const u16* A, const u16* Bm, const float* bias, float* Cf,
                  u16* Cb, int Mm, int Nn, int Kk, int flags) {
    dim3 g(Nn / 128, (Mm + 127) / 128);
    gemm_bf16<<<g, 256, 0, stream>>>(A, Bm, bias, Cf, Cb, Mm, Nn, Kk, flags);
  };

  // weight / input conversions
  cvt(W_mem, Wmem_b, E * E);
  cvt(W_text, Wtext_b, E * E);
  cvt(W_pair, Wpair_b, 2 * E * E);
  cvt(Wq, Wq_b, 4 * E * E);  cvt(Wk, Wk_b, 4 * E * E);
  cvt(Wv, Wv_b, 4 * E * E);  cvt(Wo, Wo_b, 4 * E * E);
  cvt(W_ff1, Wf1_b, 4 * E * FF);
  cvt(W_ff2, Wf2_b, 4 * FF * E);
  cvt(memory, mem_b, M * E);
  cvt(mtext, textin_b, 32 * E);

  // memory / text projections (only the consumed precision is stored)
  gemm(mem_b, Wmem_b, b_mem, nullptr, memp_b, M, E, E, 1);
  gemm(textin_b, Wtext_b, b_text, text_f, nullptr, 32, E, E, 1);

  // pair queries
  build_pairs<<<P, 256, 0, stream>>>(hidden, refs, pair_b, refpts);
  gemm(pair_b, Wpair_b, b_pair, q_f, nullptr, P, E, 2 * E, 1);
  add_ref_cvt<<<P, 256, 0, stream>>>(q_f, refpts, W_ref, q_b);

  const float att_scale = 0.17677669529663687f;   // 1/sqrt(32)
  for (int l = 0; l < 4; ++l) {
    gemm(q_b,    Wq_b + (size_t)l * E * E, bq + l * E, nullptr, qh_b, P, E, E, 1);
    gemm(memp_b, Wk_b + (size_t)l * E * E, bk + l * E, nullptr, K_b, M, E, E, 1);
    gemm(memp_b, Wv_b + (size_t)l * E * E, bv + l * E, nullptr, Vt_b, M, E, E,
         1 | 4);  // bias + transposed bf16 store -> Vt[E][M]
    attention_fa<<<P / 16, 256, 0, stream>>>(qh_b, K_b, Vt_b, mmask, attn_b,
                                             att_scale, M);
    gemm(attn_b, Wo_b + (size_t)l * E * E, bo + l * E, tmp2_f, nullptr, P, E, E, 1);
    ln_residual<<<P, 256, 0, stream>>>(q_f, tmp2_f, ln1_g + l * E, ln1_b + l * E,
                                       q_f, q_b);
    gemm(q_b,  Wf1_b + (size_t)l * E * FF, b_ff1 + l * FF, nullptr, f1_b, P, FF, E, 3);
    gemm(f1_b, Wf2_b + (size_t)l * FF * E, b_ff2 + l * E, tmp2_f, nullptr, P, E, FF, 1);
    ln_residual<<<P, 256, 0, stream>>>(q_f, tmp2_f, ln2_g + l * E, ln2_b + l * E,
                                       q_f, q_b);
  }

  // contrastive scores + symmetric scatter
  const int outN = 64 * 64 * 32;
  zero_f32<<<(outN + 255) / 256, 256, 0, stream>>>((float*)d_out, outN);
  scores_scatter<<<P, 32, 0, stream>>>(q_f, text_f, tmask, log_scale, (float*)d_out);
}
